// TransformerNetwork_13228499272204
// MI455X (gfx1250) — matrix operand order repaired and checked by reading, not compiled, not run on hardware
//
#include <hip/hip_runtime.h>
#include <cstdint>
#include <cstddef>
#include <cmath>

// ---------------- problem dimensions ----------------
#define NB   32      // batch
#define NS   256     // sequence / patch length
#define ND   900     // d_model
#define NDP  928     // d_model padded to mult of 32
#define NDFF 512     // LSTM width
#define NH3  1536    // 3*DFF
#define NNH  10      // heads
#define NDK  90      // head dim
#define NDKP 96      // head dim padded
#define ND5  4500    // 5*D
#define ND5P 4512    // padded
#define NM   8192    // NB*NS rows

typedef __attribute__((ext_vector_type(16))) __bf16 v16bf;
typedef __attribute__((ext_vector_type(8)))  float  v8f;
typedef __attribute__((ext_vector_type(4)))  int    v4i;

#ifndef __has_builtin
#define __has_builtin(x) 0
#endif
#if __has_builtin(__builtin_amdgcn_global_load_async_to_lds_b128)
#define HAVE_ASYNC_LDS 1
#else
#define HAVE_ASYNC_LDS 0
#endif

// ---------------- small device helpers ----------------
__device__ inline unsigned short f2bf(float f) {
  unsigned u = __float_as_uint(f);
  unsigned r = ((u >> 16) & 1u) + 0x7fffu;   // round-to-nearest-even
  return (unsigned short)((u + r) >> 16);
}
__device__ inline float bf2f(unsigned short h) {
  return __uint_as_float(((unsigned)h) << 16);
}
// Build a 16xbf16 WMMA fragment from two 16B LDS chunks (ds_load_b128 x2).
__device__ inline v16bf load_frag(const unsigned short* p, int off_elems) {
  union { uint4 u[2]; v16bf v; } f;
  f.u[0] = *(const uint4*)p;
  f.u[1] = *(const uint4*)(p + off_elems);
  return f.v;
}
__device__ inline unsigned wanghash(unsigned x) {
  x = (x ^ 61u) ^ (x >> 16); x *= 9u; x ^= x >> 4; x *= 0x27d4eb2du; x ^= x >> 15;
  return x;
}

#if HAVE_ASYNC_LDS
// async global->LDS 16B copy (VGPR-bypass path, tracked with ASYNCcnt)
typedef __attribute__((address_space(1))) v4i* gp_v4i;   // global int4*
typedef __attribute__((address_space(3))) v4i* lp_v4i;   // LDS int4*
__device__ inline void async_g2l_b128(const void* g, void* l) {
  __builtin_amdgcn_global_load_async_to_lds_b128((gp_v4i)g, (lp_v4i)l, 0, 0);
}
__device__ inline void wait_async0() {
#if __has_builtin(__builtin_amdgcn_s_wait_asynccnt)
  __builtin_amdgcn_s_wait_asynccnt(0);
#else
  asm volatile("s_wait_asynccnt 0x0" ::: "memory");
#endif
}
#endif

// ---------------- fp32 -> bf16 convert with K padding ----------------
__global__ void cvt_pad_kernel(const float* __restrict__ src, int lds_,
                               unsigned short* __restrict__ dst, int ldd,
                               int M, int K, int Kpad) {
  size_t idx = (size_t)blockIdx.x * blockDim.x + threadIdx.x;
  size_t total = (size_t)M * Kpad;
  if (idx >= total) return;
  int r = (int)(idx / Kpad), c = (int)(idx % Kpad);
  dst[(size_t)r * ldd + c] = (c < K) ? f2bf(src[(size_t)r * lds_ + c])
                                     : (unsigned short)0;
}

// ---------------- WMMA GEMM:  out = act(A[M,K] * W[N,K]^T + bias (+resid)) ----
// 128x128 block tile, 8 waves (4Mx2N), each wave 32x64 via 2x4 16x16x32 WMMAs.
// Double-buffered LDS; async global->LDS staging overlaps next tile with WMMAs.
__global__ __launch_bounds__(256)
void gemm_wmma_kernel(const unsigned short* __restrict__ A, int lda,
                      const unsigned short* __restrict__ W, int ldb,
                      const float* __restrict__ bias,
                      const float* __restrict__ resid, int ldr,
                      float* __restrict__ outF, int ldo,
                      unsigned short* __restrict__ outB, int ldob,
                      int M, int N, int Npad, int K, int act) {
  __shared__ unsigned short As[2 * 128 * 40];
  __shared__ unsigned short Bs[2 * 128 * 40];
  const int tid  = threadIdx.x;
  const int lane = tid & 31, wid = tid >> 5;
  const int half = lane >> 4, lr = lane & 15;
  const int wRow = (wid >> 1) * 32;   // wave M offset inside tile
  const int wCol = (wid & 1) * 64;    // wave N offset inside tile
  const int tileM = blockIdx.y * 128;
  const int tileN = blockIdx.x * 128;

  v8f acc[2][4];
  const v8f vz = {0.f,0.f,0.f,0.f,0.f,0.f,0.f,0.f};
  for (int m = 0; m < 2; ++m) for (int n = 0; n < 4; ++n) acc[m][n] = vz;

  auto stage = [&](int buf, int k0) {
    unsigned short* Asb = &As[buf * 128 * 40];
    unsigned short* Bsb = &Bs[buf * 128 * 40];
    for (int t = 0; t < 2; ++t) {
      int chunk = tid + t * 256;            // 0..511
      int row = chunk >> 2;
      int c8  = (chunk & 3) << 3;
      int gr = tileM + row;
      int gc = tileN + row;
#if HAVE_ASYNC_LDS
      if (gr < M) async_g2l_b128(A + (size_t)gr * lda + k0 + c8, &Asb[row * 40 + c8]);
      else        *(uint4*)&Asb[row * 40 + c8] = make_uint4(0, 0, 0, 0);
      if (gc < N) async_g2l_b128(W + (size_t)gc * ldb + k0 + c8, &Bsb[row * 40 + c8]);
      else        *(uint4*)&Bsb[row * 40 + c8] = make_uint4(0, 0, 0, 0);
#else
      uint4 av = make_uint4(0, 0, 0, 0);
      if (gr < M) av = *(const uint4*)(A + (size_t)gr * lda + k0 + c8);
      *(uint4*)&Asb[row * 40 + c8] = av;
      uint4 bv = make_uint4(0, 0, 0, 0);
      if (gc < N) bv = *(const uint4*)(W + (size_t)gc * ldb + k0 + c8);
      *(uint4*)&Bsb[row * 40 + c8] = bv;
#endif
    }
  };

  auto compute = [&](int buf) {
    const unsigned short* Asb = &As[buf * 128 * 40];
    const unsigned short* Bsb = &Bs[buf * 128 * 40];
    v16bf af[2], bfr[4];
    for (int m = 0; m < 2; ++m)   // A frag: lane half selects K 0..7/16..23 vs 8..15/24..31
      af[m] = load_frag(&Asb[(wRow + m * 16 + lr) * 40 + half * 8], 16);
    for (int n = 0; n < 4; ++n)   // B frag: lane half selects K 0..7/16..23 vs 8..15/24..31
      bfr[n] = load_frag(&Bsb[(wCol + n * 16 + lr) * 40 + half * 8], 16);
    for (int m = 0; m < 2; ++m)
      for (int n = 0; n < 4; ++n)
        acc[m][n] = __builtin_amdgcn_wmma_f32_16x16x32_bf16(
            false, af[m], false, bfr[n], (short)0, acc[m][n], false, false);
  };

  const int nk = K >> 5;          // K is always a multiple of 32
  stage(0, 0);
  for (int ki = 0; ki < nk; ++ki) {
#if HAVE_ASYNC_LDS
    wait_async0();                // this wave's pending async loads done
#endif
    __syncthreads();              // all waves' staging of buf visible
    if (ki + 1 < nk) stage((ki + 1) & 1, (ki + 1) << 5);  // prefetch next tile
    compute(ki & 1);              // 8 WMMAs overlap with in-flight async loads
    __syncthreads();              // reads of buf done before it is re-staged
  }

  // epilogue: C layout -> lane holds rows (half*8 + j), col = lane&15
  for (int m = 0; m < 2; ++m)
    for (int n = 0; n < 4; ++n)
      for (int j = 0; j < 8; ++j) {
        int gr = tileM + wRow + m * 16 + half * 8 + j;
        int gc = tileN + wCol + n * 16 + lr;
        if (gr >= M) continue;
        if (gc < N) {
          float v = acc[m][n][j];
          if (bias)  v += bias[gc];
          if (resid) v += resid[(size_t)gr * ldr + gc];
          if (act == 1) v = 0.5f * v * (1.0f + erff(v * 0.70710678118f)); // exact GELU
          if (outF) outF[(size_t)gr * ldo + gc] = v;
          if (outB) outB[(size_t)gr * ldob + gc] = f2bf(v);
        } else if (outB && gc < Npad) {
          outB[(size_t)gr * ldob + gc] = 0;   // zero K-padding for next GEMM
        }
      }
}

// ---------------- row LayerNorm over D=900 ----------------
__global__ __launch_bounds__(256)
void ln_kernel(const float* __restrict__ x, const float* __restrict__ g,
               const float* __restrict__ b, float* __restrict__ outF,
               unsigned short* __restrict__ outB, int ldob, int N, int Npad) {
  const int row = blockIdx.x;
  const float* xr = x + (size_t)row * N;
  __shared__ float red[256];
  float s = 0.f;
  for (int i = threadIdx.x; i < N; i += 256) s += xr[i];
  red[threadIdx.x] = s; __syncthreads();
  for (int o = 128; o > 0; o >>= 1) {
    if (threadIdx.x < o) red[threadIdx.x] += red[threadIdx.x + o];
    __syncthreads();
  }
  float mean = red[0] / N; __syncthreads();
  float v = 0.f;
  for (int i = threadIdx.x; i < N; i += 256) { float d = xr[i] - mean; v += d * d; }
  red[threadIdx.x] = v; __syncthreads();
  for (int o = 128; o > 0; o >>= 1) {
    if (threadIdx.x < o) red[threadIdx.x] += red[threadIdx.x + o];
    __syncthreads();
  }
  float rs = rsqrtf(red[0] / N + 1e-5f);
  for (int i = threadIdx.x; i < Npad; i += 256) {
    if (i < N) {
      float y = (xr[i] - mean) * rs * g[i] + b[i];
      if (outF) outF[(size_t)row * N + i] = y;
      if (outB) outB[(size_t)row * ldob + i] = f2bf(y);
    } else if (outB) {
      outB[(size_t)row * ldob + i] = 0;
    }
  }
}

// ---------------- q = qe*qh, repack [B,S,NH,DK] -> [B,NH,S,DKP] bf16 ---------
__global__ void mulpack_kernel(const float* __restrict__ xe,
                               const float* __restrict__ xh,
                               unsigned short* __restrict__ dst) {
  size_t idx = (size_t)blockIdx.x * blockDim.x + threadIdx.x;
  size_t total = (size_t)NB * NNH * NS * NDKP;
  if (idx >= total) return;
  int d = (int)(idx % NDKP);
  int s = (int)((idx / NDKP) % NS);
  int h = (int)((idx / ((size_t)NDKP * NS)) % NNH);
  int b = (int)(idx / ((size_t)NDKP * NS * NNH));
  unsigned short v = 0;
  if (d < NDK) {
    size_t off = (size_t)(b * NS + s) * ND + h * NDK + d;
    v = f2bf(xe[off] * xh[off]);
  }
  dst[idx] = v;
}

// ---------------- attention: one block per (b, h, 64-row q tile) -------------
__global__ __launch_bounds__(256)
void attention_kernel(const unsigned short* __restrict__ Q,
                      const unsigned short* __restrict__ Km,
                      const unsigned short* __restrict__ Vm,
                      float* __restrict__ out, float scale) {
  __shared__ unsigned short Qs[64 * 104];     // 64 x 96 (stride 104)
  __shared__ unsigned short KVs[256 * 104];   // K 256x96, later V^T 96x256
  __shared__ unsigned short Ps[64 * 264];     // scores / probs bf16
  const int tid  = threadIdx.x;
  const int lane = tid & 31, wid = tid >> 5;
  const int half = lane >> 4, lr = lane & 15;
  const int qt = blockIdx.x, h = blockIdx.y, b = blockIdx.z;
  const size_t bh = (size_t)(b * NNH + h) * NS * NDKP;

  for (int t = 0; t < 3; ++t) {               // stage Q tile (full, no guards)
    int chunk = tid + t * 256;                // 0..767
    int row = chunk / 12; int c8 = (chunk % 12) * 8;
#if HAVE_ASYNC_LDS
    async_g2l_b128(Q + bh + (size_t)(qt * 64 + row) * NDKP + c8, &Qs[row * 104 + c8]);
#else
    uint4 v = *(const uint4*)(Q + bh + (size_t)(qt * 64 + row) * NDKP + c8);
    *(uint4*)&Qs[row * 104 + c8] = v;
#endif
  }
  for (int t = 0; t < 12; ++t) {              // stage all K
    int chunk = tid + t * 256;                // 0..3071
    int row = chunk / 12; int c8 = (chunk % 12) * 8;
#if HAVE_ASYNC_LDS
    async_g2l_b128(Km + bh + (size_t)row * NDKP + c8, &KVs[row * 104 + c8]);
#else
    uint4 v = *(const uint4*)(Km + bh + (size_t)row * NDKP + c8);
    *(uint4*)&KVs[row * 104 + c8] = v;
#endif
  }
#if HAVE_ASYNC_LDS
  wait_async0();
#endif
  __syncthreads();

  { // scores 64x256 = Q * K^T, waves 2(M) x 4(N)
    const int wM = (wid >> 2) * 32;
    const int wN = (wid & 3) * 64;
    v8f sa[2][4];
    const v8f vz = {0.f,0.f,0.f,0.f,0.f,0.f,0.f,0.f};
    for (int m = 0; m < 2; ++m) for (int n = 0; n < 4; ++n) sa[m][n] = vz;
    for (int k0 = 0; k0 < NDKP; k0 += 32) {
      v16bf af[2], bfr[4];
      for (int m = 0; m < 2; ++m)
        af[m] = load_frag(&Qs[(wM + m * 16 + lr) * 104 + k0 + half * 8], 16);
      for (int n = 0; n < 4; ++n)
        bfr[n] = load_frag(&KVs[(wN + n * 16 + lr) * 104 + k0 + half * 8], 16);
      for (int m = 0; m < 2; ++m)
        for (int n = 0; n < 4; ++n)
          sa[m][n] = __builtin_amdgcn_wmma_f32_16x16x32_bf16(
              false, af[m], false, bfr[n], (short)0, sa[m][n], false, false);
    }
    for (int m = 0; m < 2; ++m)
      for (int n = 0; n < 4; ++n)
        for (int j = 0; j < 8; ++j) {
          int r = wM + m * 16 + half * 8 + j;
          int c = wN + n * 16 + lr;
          Ps[r * 264 + c] = f2bf(sa[m][n][j] * scale);
        }
  }
  __syncthreads();

  // softmax (threads 0..63, one row each) while all threads stage V^T
  if (tid < 64) {
    float mx = -3.0e38f;
    for (int i = 0; i < 256; ++i) mx = fmaxf(mx, bf2f(Ps[tid * 264 + i]));
    float sum = 0.f;
    for (int i = 0; i < 256; ++i) {
      float e = expf(bf2f(Ps[tid * 264 + i]) - mx);
      sum += e;
      Ps[tid * 264 + i] = f2bf(e);
    }
    float inv = 1.0f / sum;
    for (int i = 0; i < 256; ++i)
      Ps[tid * 264 + i] = f2bf(bf2f(Ps[tid * 264 + i]) * inv);
  }
  for (int t = 0; t < 96; ++t) {              // V^T into KVs (stride 264)
    int e = tid + t * 256;                    // 0..24575
    int d = e % NDKP; int kk = e / NDKP;
    KVs[d * 264 + kk] = Vm[bh + (size_t)kk * NDKP + d];
  }
  __syncthreads();

  { // attn 64x96 = P * V, waves 4(M) x 2(N)
    const int wM = (wid >> 1) * 16;
    const int wN = (wid & 1) * 48;
    v8f oa[3];
    const v8f vz = {0.f,0.f,0.f,0.f,0.f,0.f,0.f,0.f};
    for (int n = 0; n < 3; ++n) oa[n] = vz;
    for (int k0 = 0; k0 < 256; k0 += 32) {
      v16bf af = load_frag(&Ps[(wM + lr) * 264 + k0 + half * 8], 16);
      for (int n = 0; n < 3; ++n) {
        v16bf bfr = load_frag(&KVs[(wN + n * 16 + lr) * 264 + k0 + half * 8], 16);
        oa[n] = __builtin_amdgcn_wmma_f32_16x16x32_bf16(
            false, af, false, bfr, (short)0, oa[n], false, false);
      }
    }
    for (int n = 0; n < 3; ++n)
      for (int j = 0; j < 8; ++j) {
        int rl = wM + half * 8 + j;
        int d  = wN + n * 16 + lr;
        if (d < NDK) {
          int grow = b * NS + qt * 64 + rl;
          out[(size_t)grow * ND + h * NDK + d] = oa[n][j];
        }
      }
  }
}

// ---------------- Z1 = state + attn ----------------
__global__ void add_kernel(const float* __restrict__ a, const float* __restrict__ b,
                           float* __restrict__ o, size_t n) {
  size_t i = (size_t)blockIdx.x * blockDim.x + threadIdx.x;
  if (i < n) o[i] = a[i] + b[i];
}

// ---------------- LSTM pointwise (stabilized exp gates + noise) --------------
__global__ void lstm_pointwise_kernel(const float* __restrict__ GI, const float* __restrict__ GF,
                                      const float* __restrict__ GO, const float* __restrict__ GZ,
                                      const float* __restrict__ Ci, const float* __restrict__ Mi,
                                      const float* __restrict__ Ni,
                                      float* __restrict__ Co, float* __restrict__ Mo,
                                      float* __restrict__ Ho, float* __restrict__ No,
                                      unsigned short* __restrict__ hslab, int hstride,
                                      unsigned seed) {
  size_t i = (size_t)blockIdx.x * blockDim.x + threadIdx.x;
  if (i >= (size_t)NM * NDFF) return;
  float it = GI[i], ft = GF[i], ot = GO[i], zt = GZ[i];
  float ci = Ci[i], mi = Mi[i], ni = Ni[i];
  float mt = fmaxf(ft + mi, it);
  float Ig = expf(it - mt);
  float Fg = expf(ft + mi - mt);
  unsigned h1 = wanghash((unsigned)i * 2u + 1u + seed);
  unsigned h2 = wanghash((unsigned)i * 2u + 2u + seed);
  unsigned h3 = wanghash(h1 ^ 0x85ebca6bu);
  unsigned h4 = wanghash(h2 ^ 0xc2b2ae35u);
  float u1 = ((h1 >> 8) + 1u) * (1.0f / 16777217.0f);
  float u2 = (h3 >> 8) * (1.0f / 16777216.0f);
  float u3 = ((h2 >> 8) + 1u) * (1.0f / 16777217.0f);
  float u4 = (h4 >> 8) * (1.0f / 16777216.0f);
  float n1 = sqrtf(-2.0f * logf(u1)) * cosf(6.28318531f * u2);
  float n2 = sqrtf(-2.0f * logf(u3)) * cosf(6.28318531f * u4);
  float Og = 1.0f / (1.0f + expf(-(ot + 0.01f * n1)));
  float Zg = tanhf(zt + 0.01f * n2);
  float C  = ci * Fg + Ig * Zg;
  float Hn = Og * (C / ni);
  float Nn = Fg * ni + Ig;
  Co[i] = C; Mo[i] = mt; Ho[i] = Hn; No[i] = Nn;
  int r = (int)(i >> 9), c = (int)(i & 511);
  hslab[(size_t)r * hstride + c] = f2bf(Hn);   // refresh H-concat slab for next TB
}

// ======================= host driver =======================
extern "C" void kernel_launch(void* const* d_in, const int* in_sizes, int n_in,
                              void* d_out, int out_size, void* d_ws, size_t ws_size,
                              hipStream_t stream) {
  (void)in_sizes; (void)n_in; (void)out_size; (void)ws_size;
  auto F = [&](int i) { return (const float*)d_in[i]; };

  // ---- workspace carve ----
  char* wsp = (char*)d_ws;
  auto carve = [&](size_t bytes) -> char* {
    char* p = wsp; wsp += (bytes + 255) & ~(size_t)255; return p;
  };
  unsigned short* WSLAB = (unsigned short*)carve((size_t)ND5 * NDP * 2);   // weight bf16 slab
  unsigned short* XBUF  = (unsigned short*)carve((size_t)NM * NDP * 2);    // state/Z bf16
  unsigned short* XBUF2 = (unsigned short*)carve((size_t)NM * NDP * 2);    // LN(Z1) bf16
  unsigned short* HCAT  = (unsigned short*)carve((size_t)NM * NH3 * 2);    // [H1|H2|H3] bf16
  unsigned short* MBUF  = (unsigned short*)carve((size_t)NM * ND5P * 2);   // gelu(W1 x) bf16
  float* T0 = (float*)carve((size_t)NM * ND * 4);
  float* T1 = (float*)carve((size_t)NM * ND * 4);
  float* ZA = (float*)carve((size_t)NM * ND * 4);
  float* ZB = (float*)carve((size_t)NM * ND * 4);
  unsigned short* QPK = (unsigned short*)carve((size_t)NB * NNH * NS * NDKP * 2);
  unsigned short* KPK = (unsigned short*)carve((size_t)NB * NNH * NS * NDKP * 2);
  unsigned short* VPK = (unsigned short*)carve((size_t)NB * NNH * NS * NDKP * 2);
  float* G[4];
  for (int i = 0; i < 4; ++i) G[i] = (float*)carve((size_t)NM * NDFF * 4);

  // ---- launch helpers ----
  auto cvt = [&](const float* src, int M, int K, int lds_, unsigned short* dst,
                 int ldd, int Kpad) {
    size_t total = (size_t)M * Kpad;
    cvt_pad_kernel<<<(unsigned)((total + 255) / 256), 256, 0, stream>>>(
        src, lds_, dst, ldd, M, K, Kpad);
  };
  auto gemm = [&](const unsigned short* A, int lda, const unsigned short* W, int ldb,
                  const float* bias, const float* resid, int ldr,
                  float* outF, int ldo, unsigned short* outB, int ldob,
                  int M, int N, int Npad, int K, int act) {
    dim3 g((N + 127) / 128, (M + 127) / 128);
    gemm_wmma_kernel<<<g, 256, 0, stream>>>(A, lda, W, ldb, bias, resid, ldr,
                                            outF, ldo, outB, ldob, M, N, Npad, K, act);
  };
  auto ln = [&](const float* x, const float* g, const float* b, float* outF,
                unsigned short* outB, int ldob, int Npad) {
    ln_kernel<<<NM, 256, 0, stream>>>(x, g, b, outF, outB, ldob, ND, Npad);
  };

  // ---- init: bf16 state + H-concat slabs ----
  cvt(F(0),  NM, ND,   ND,   XBUF,            NDP, NDP);
  cvt(F(3),  NM, NDFF, NDFF, HCAT + 0 * NDFF, NH3, NDFF);
  cvt(F(7),  NM, NDFF, NDFF, HCAT + 1 * NDFF, NH3, NDFF);
  cvt(F(11), NM, NDFF, NDFF, HCAT + 2 * NDFF, NH3, NDFF);

  const float att_scale = 1.0f / sqrtf((float)NDK);

  // ---- TransformerBlock: param base pb ----
  auto run_tb = [&](int pb, const float* stateF, float* outZF) {
    unsigned short* PK[3] = {QPK, KPK, VPK};
    for (int s = 0; s < 3; ++s) {
      // qe/ke/ve = LN(state @ W^T + b)
      cvt(F(pb + s), ND, ND, ND, WSLAB, NDP, NDP);
      gemm(XBUF, NDP, WSLAB, NDP, F(pb + 10 + s), nullptr, 0,
           T0, ND, nullptr, 0, NM, ND, ND, NDP, 0);
      ln(T0, F(pb + 16 + 2 * s), F(pb + 17 + 2 * s), T0, nullptr, 0, ND);
      // qh/kh/vh = LN(H @ WC^T + bC)
      cvt(F(pb + 3 + s), ND, NH3, NH3, WSLAB, NH3, NH3);
      gemm(HCAT, NH3, WSLAB, NH3, F(pb + 13 + s), nullptr, 0,
           T1, ND, nullptr, 0, NM, ND, ND, NH3, 0);
      ln(T1, F(pb + 22 + 2 * s), F(pb + 23 + 2 * s), T1, nullptr, 0, ND);
      size_t tot = (size_t)NB * NNH * NS * NDKP;
      mulpack_kernel<<<(unsigned)((tot + 255) / 256), 256, 0, stream>>>(T0, T1, PK[s]);
    }
    attention_kernel<<<dim3(4, NNH, NB), 256, 0, stream>>>(QPK, KPK, VPK, T0, att_scale);
    size_t nz = (size_t)NM * ND;
    add_kernel<<<(unsigned)((nz + 255) / 256), 256, 0, stream>>>(stateF, T0, outZF, nz);
    // Z2n = LN(Z1) -> bf16
    ln(outZF, F(pb + 28), F(pb + 29), nullptr, XBUF2, NDP, NDP);
    // Z3 = gelu(Z2n @ W1^T + b1) -> bf16 only
    cvt(F(pb + 6), ND5, ND, ND, WSLAB, NDP, NDP);
    gemm(XBUF2, NDP, WSLAB, NDP, F(pb + 8), nullptr, 0,
         nullptr, 0, MBUF, ND5P, NM, ND5, ND5P, NDP, 1);
    // Z = Z1 + Z3 @ W2^T + b2  -> fp32 (outZF) and bf16 (XBUF for next stage)
    cvt(F(pb + 7), ND, ND5, ND5, WSLAB, ND5P, ND5P);
    gemm(MBUF, ND5P, WSLAB, ND5P, F(pb + 9), outZF, ND,
         outZF, ND, XBUF, NDP, NM, ND, NDP, ND5P, 0);
  };

  // ---- LSTM: param base lb, layer index ----
  auto run_lstm = [&](int lb, int layer) {
    for (int g4 = 0; g4 < 4; ++g4) {
      // W-gate: H_layer @ Wg^T + bWg
      cvt(F(lb + 2 * g4), NDFF, NDFF, NDFF, WSLAB, NDFF, NDFF);
      gemm(HCAT + layer * NDFF, NH3, WSLAB, NDFF, F(lb + 2 * g4 + 1), nullptr, 0,
           G[g4], NDFF, nullptr, 0, NM, NDFF, NDFF, NDFF, 0);
      // + Z @ Rg^T + bRg (accumulated via residual path)
      cvt(F(lb + 8 + 2 * g4), NDFF, ND, ND, WSLAB, NDP, NDP);
      gemm(XBUF, NDP, WSLAB, NDP, F(lb + 8 + 2 * g4 + 1), G[g4], NDFF,
           G[g4], NDFF, nullptr, 0, NM, NDFF, NDFF, NDP, 0);
    }
    float* ob = (float*)d_out;
    size_t zoff = (size_t)NM * ND;
    size_t st   = (size_t)NM * NDFF;
    float* Co = ob + zoff + (size_t)(4 * layer + 0) * st;
    float* Mo = ob + zoff + (size_t)(4 * layer + 1) * st;
    float* Ho = ob + zoff + (size_t)(4 * layer + 2) * st;
    float* No = ob + zoff + (size_t)(4 * layer + 3) * st;
    int ci = 1 + 4 * layer;   // C_i, M_i, H_i, N_i input indices
    size_t n = (size_t)NM * NDFF;
    lstm_pointwise_kernel<<<(unsigned)((n + 255) / 256), 256, 0, stream>>>(
        G[0], G[1], G[2], G[3], F(ci), F(ci + 1), F(ci + 3),
        Co, Mo, Ho, No, HCAT + layer * NDFF, NH3, 0x9E3779B9u * (unsigned)(layer + 1));
  };

  // ---- full pipeline ----
  run_tb(13 + 0 * 30, F(0), ZA);             run_lstm(103 + 0 * 16, 0);
  run_tb(13 + 1 * 30, ZA,   ZB);             run_lstm(103 + 1 * 16, 1);
  run_tb(13 + 2 * 30, ZB,   (float*)d_out);  run_lstm(103 + 2 * 16, 2);
}
